// QGSNET_semseg_s3dis_61890478735432
// MI455X (gfx1250) — matrix operand order repaired and checked by reading, not compiled, not run on hardware
//
#include <hip/hip_runtime.h>
#include <hip/hip_bf16.h>
#include <cstdint>

// ---------------------------------------------------------------------------
// QGSNET semseg on MI455X (gfx1250): all channel-matmuls via
// v_wmma_f32_16x16x32_bf16 (wave32), irregular ops (knn/fps/eig/interp) as
// VALU kernels.  fp32 master precision, bf16 GEMM staging.
// ---------------------------------------------------------------------------

typedef __bf16 bf16_t;
typedef __attribute__((ext_vector_type(16))) __bf16 v16bf;
typedef __attribute__((ext_vector_type(8)))  __bf16 v8bf;
typedef __attribute__((ext_vector_type(8)))  float  v8f;

#define DEV __device__ __forceinline__

constexpr int   BB  = 4;
constexpr int   N1  = 4096;
constexpr int   N2  = 2048;
constexpr int   N3  = 1024;
constexpr int   KNN = 20;
constexpr float EPS_BN = 1e-5f;
constexpr float NEG    = 0.2f;

// ======================= WMMA fragment loaders =============================
// A: row-major [16 x Kc] tile base.  K order matches load_b_frag (contiguous):
// lanes 0-15 hold K 0..15; lanes 16-31 hold K 16..31.
DEV v16bf load_a_frag(const bf16_t* __restrict__ A, int ldk, int lane, int kk) {
    int row = lane & 15;
    int kh  = lane >> 4;
    const bf16_t* p = A + (size_t)row * ldk + kk;
    v8bf lo = *(const v8bf*)(p + kh * 16);
    v8bf hi = *(const v8bf*)(p + kh * 16 + 8);
    v16bf a;
#pragma unroll
    for (int i = 0; i < 8; ++i) { a[i] = lo[i]; a[i + 8] = hi[i]; }
    return a;
}

// B: we pass W row-major [Nout x Kc] == B^T, so B[k][n] = W[n][k].
// 32x16 B layout: column = lane&15; lanes 0-15 K 0..15, lanes 16-31 K 16..31.
// -> one contiguous 32-byte load per lane.
DEV v16bf load_b_frag(const bf16_t* __restrict__ W, int ldk, int lane, int kk) {
    int col = lane & 15;
    int kh  = lane >> 4;
    return *(const v16bf*)(W + (size_t)col * ldk + kk + kh * 16);
}

// ======================= GEMM kernels ======================================
// Y[M,Nout] = A[M,Kc] * W[Nout,Kc]^T ; optional per-channel sum/sumsq stats.
template <bool STATS>
__global__ __launch_bounds__(32) void qgs_wmma_gemm(
    const bf16_t* __restrict__ A, const bf16_t* __restrict__ W,
    float* __restrict__ Y, float* __restrict__ stats,
    int M, int Nout, int Kc)
{
    int mt = blockIdx.x, nt = blockIdx.y;
    int lane = threadIdx.x;
    const bf16_t* Ab = A + (size_t)mt * 16 * Kc;
    const bf16_t* Wb = W + (size_t)nt * 16 * Kc;
    v8f acc = {};
    for (int kk = 0; kk < Kc; kk += 32) {
        v16bf a = load_a_frag(Ab, Kc, lane, kk);
        v16bf b = load_b_frag(Wb, Kc, lane, kk);
        acc = __builtin_amdgcn_wmma_f32_16x16x32_bf16(false, a, false, b,
                                                      (short)0, acc, false, false);
    }
    int col = lane & 15;
    int rh  = (lane >> 4) * 8;
    int gn  = nt * 16 + col;
    float s = 0.f, s2 = 0.f;
#pragma unroll
    for (int i = 0; i < 8; ++i) {
        float v = acc[i];
        int gm = mt * 16 + rh + i;
        Y[(size_t)gm * Nout + gn] = v;
        s += v; s2 += v * v;
    }
    if (STATS) {
        atomicAdd(&stats[gn], s);
        atomicAdd(&stats[1024 + gn], s2);
    }
}

// QGSCM fused: t = (E*Wr^T) * (E*Wg^T + bg) + (E2*Wb^T), plus BN stats.
__global__ __launch_bounds__(32) void qgs_wmma_qgscm(
    const bf16_t* __restrict__ E, const bf16_t* __restrict__ E2,
    const bf16_t* __restrict__ Wr, const bf16_t* __restrict__ Wg,
    const bf16_t* __restrict__ Wb_, const float* __restrict__ bg,
    float* __restrict__ T, float* __restrict__ stats,
    int M, int Cout, int Kc)
{
    int mt = blockIdx.x, nt = blockIdx.y;
    int lane = threadIdx.x;
    const bf16_t* Eb  = E  + (size_t)mt * 16 * Kc;
    const bf16_t* E2b = E2 + (size_t)mt * 16 * Kc;
    const bf16_t* Wrb = Wr  + (size_t)nt * 16 * Kc;
    const bf16_t* Wgb = Wg  + (size_t)nt * 16 * Kc;
    const bf16_t* Wbb = Wb_ + (size_t)nt * 16 * Kc;
    v8f ar = {}, ag = {}, ab = {};
    for (int kk = 0; kk < Kc; kk += 32) {
        v16bf ea  = load_a_frag(Eb,  Kc, lane, kk);
        v16bf e2a = load_a_frag(E2b, Kc, lane, kk);
        v16bf br = load_b_frag(Wrb, Kc, lane, kk);
        v16bf bg_ = load_b_frag(Wgb, Kc, lane, kk);
        v16bf bb = load_b_frag(Wbb, Kc, lane, kk);
        ar = __builtin_amdgcn_wmma_f32_16x16x32_bf16(false, ea,  false, br,  (short)0, ar, false, false);
        ag = __builtin_amdgcn_wmma_f32_16x16x32_bf16(false, ea,  false, bg_, (short)0, ag, false, false);
        ab = __builtin_amdgcn_wmma_f32_16x16x32_bf16(false, e2a, false, bb,  (short)0, ab, false, false);
    }
    int col = lane & 15;
    int rh  = (lane >> 4) * 8;
    int gn  = nt * 16 + col;
    float bias = bg[gn];
    float s = 0.f, s2 = 0.f;
#pragma unroll
    for (int i = 0; i < 8; ++i) {
        float t = ar[i] * (ag[i] + bias) + ab[i];
        int gm = mt * 16 + rh + i;
        T[(size_t)gm * Cout + gn] = t;
        s += t; s2 += t * t;
    }
    atomicAdd(&stats[gn], s);
    atomicAdd(&stats[1024 + gn], s2);
}

// ======================= irregular / elementwise kernels ===================
__global__ void qgs_extract_pts(const float* __restrict__ x, float* __restrict__ pts) {
    size_t t = (size_t)blockIdx.x * blockDim.x + threadIdx.x;
    if (t >= (size_t)BB * N1 * 3) return;
    int d = t % 3;
    size_t bn = t / 3;
    int b = bn / N1, n = bn % N1;
    pts[t] = x[((size_t)b * 9 + d) * N1 + n];   // x is [B,9,N]; first 3 channels
}

// brute-force top-20 nearest (incl self), 3-dim points, one thread per query
__global__ void qgs_knn(const float* __restrict__ pts, int* __restrict__ idx, int Np) {
    int t = blockIdx.x * blockDim.x + threadIdx.x;
    if (t >= BB * Np) return;
    int b = t / Np, n = t - b * Np;
    const float* base = pts + (size_t)b * Np * 3;
    float px = base[n * 3 + 0], py = base[n * 3 + 1], pz = base[n * 3 + 2];
    float bd[KNN]; int bi[KNN];
#pragma unroll
    for (int j = 0; j < KNN; ++j) { bd[j] = 3.0e38f; bi[j] = 0; }
    for (int m = 0; m < Np; ++m) {
        float dx = base[m * 3 + 0] - px;
        float dy = base[m * 3 + 1] - py;
        float dz = base[m * 3 + 2] - pz;
        float d2 = dx * dx + dy * dy + dz * dz;
        if (d2 < bd[KNN - 1]) {
            bd[KNN - 1] = d2; bi[KNN - 1] = m;
            for (int j = KNN - 1; j > 0; --j) {
                if (bd[j] < bd[j - 1]) {
                    float td = bd[j]; bd[j] = bd[j - 1]; bd[j - 1] = td;
                    int   ti = bi[j]; bi[j] = bi[j - 1]; bi[j - 1] = ti;
                } else break;
            }
        }
    }
    int* o = idx + (size_t)t * KNN;
    for (int j = 0; j < KNN; ++j) o[j] = bi[j];
}

DEV void sym3eig(float a00, float a01, float a02, float a11, float a12, float a22,
                 float& e0, float& e1, float& e2) { // ascending
    float p1 = a01 * a01 + a02 * a02 + a12 * a12;
    float q  = (a00 + a11 + a22) * (1.f / 3.f);
    float p2 = (a00 - q) * (a00 - q) + (a11 - q) * (a11 - q) + (a22 - q) * (a22 - q) + 2.f * p1;
    float p  = sqrtf(p2 * (1.f / 6.f));
    if (p < 1e-20f) { e0 = e1 = e2 = q; return; }
    float inv = 1.f / p;
    float b00 = (a00 - q) * inv, b11 = (a11 - q) * inv, b22 = (a22 - q) * inv;
    float b01 = a01 * inv, b02 = a02 * inv, b12 = a12 * inv;
    float detB = b00 * (b11 * b22 - b12 * b12) - b01 * (b01 * b22 - b12 * b02)
               + b02 * (b01 * b12 - b11 * b02);
    float r = fminf(1.f, fmaxf(-1.f, 0.5f * detB));
    float phi = acosf(r) * (1.f / 3.f);
    float big = q + 2.f * p * cosf(phi);
    float sml = q + 2.f * p * cosf(phi + 2.0943951023931953f);
    e0 = sml; e2 = big; e1 = 3.f * q - big - sml;
}

__global__ void qgs_cov_eig(const float* __restrict__ pts, const int* __restrict__ idx,
                            float* __restrict__ eig, int Np) {
    int t = blockIdx.x * blockDim.x + threadIdx.x;
    if (t >= BB * Np) return;
    int b = t / Np;
    const float* base = pts + (size_t)b * Np * 3;
    const int* id = idx + (size_t)t * KNN;
    float mx = 0.f, my = 0.f, mz = 0.f;
    for (int k = 0; k < KNN; ++k) {
        int m = id[k];
        mx += base[m * 3 + 0]; my += base[m * 3 + 1]; mz += base[m * 3 + 2];
    }
    mx *= (1.f / KNN); my *= (1.f / KNN); mz *= (1.f / KNN);
    float c00 = 0, c01 = 0, c02 = 0, c11 = 0, c12 = 0, c22 = 0;
    for (int k = 0; k < KNN; ++k) {
        int m = id[k];
        float dx = base[m * 3 + 0] - mx;
        float dy = base[m * 3 + 1] - my;
        float dz = base[m * 3 + 2] - mz;
        c00 += dx * dx; c01 += dx * dy; c02 += dx * dz;
        c11 += dy * dy; c12 += dy * dz; c22 += dz * dz;
    }
    float s = 1.f / KNN;
    float e0, e1, e2;
    sym3eig(c00 * s, c01 * s, c02 * s, c11 * s, c12 * s, c22 * s, e0, e1, e2);
    eig[(size_t)t * 3 + 0] = e0;
    eig[(size_t)t * 3 + 1] = e1;
    eig[(size_t)t * 3 + 2] = e2;
}

// layer-1 edge features: [dpts_eu, deig_eu, dpts_ei, deig_ei, dist] -> 13ch pad 32
__global__ void qgs_build_edge1(const float* __restrict__ pts, const float* __restrict__ eig,
                                const int* __restrict__ ieu, const int* __restrict__ iei,
                                bf16_t* __restrict__ E, bf16_t* __restrict__ E2, int Np) {
    size_t t = (size_t)blockIdx.x * blockDim.x + threadIdx.x;
    if (t >= (size_t)BB * Np * KNN) return;
    int k = t % KNN;
    size_t bn = t / KNN;
    int b = bn / Np, n = bn % Np;
    const float* pb = pts + (size_t)b * Np * 3;
    const float* eb = eig + (size_t)b * Np * 3;
    int meu = ieu[bn * KNN + k], mei = iei[bn * KNN + k];
    float v[13];
#pragma unroll
    for (int d = 0; d < 3; ++d) {
        v[d]     = pb[meu * 3 + d] - pb[n * 3 + d];
        v[3 + d] = eb[meu * 3 + d] - eb[n * 3 + d];
        v[6 + d] = pb[mei * 3 + d] - pb[n * 3 + d];
        v[9 + d] = eb[mei * 3 + d] - eb[n * 3 + d];
    }
    v[12] = sqrtf(v[0] * v[0] + v[1] * v[1] + v[2] * v[2]);
    bf16_t* er  = E  + t * 32;
    bf16_t* e2r = E2 + t * 32;
#pragma unroll
    for (int c = 0; c < 13; ++c) { er[c] = (bf16_t)v[c]; e2r[c] = (bf16_t)(v[c] * v[c]); }
#pragma unroll
    for (int c = 13; c < 32; ++c) { er[c] = (bf16_t)0.f; e2r[c] = (bf16_t)0.f; }
}

// layers 2/3 edge features: [f(eu)-f, f, f(ei)-f, f] -> 4C channels
__global__ void qgs_build_edge_group(const float* __restrict__ f,
                                     const int* __restrict__ ieu, const int* __restrict__ iei,
                                     bf16_t* __restrict__ E, bf16_t* __restrict__ E2,
                                     int Np, int C) {
    size_t t = (size_t)blockIdx.x * blockDim.x + threadIdx.x;
    if (t >= (size_t)BB * Np * KNN) return;
    int k = t % KNN;
    size_t bn = t / KNN;
    int b = bn / Np, n = bn % Np;
    const float* fb = f + (size_t)b * Np * C;
    const float* fc = fb + (size_t)n * C;
    const float* fe = fb + (size_t)ieu[bn * KNN + k] * C;
    const float* fi = fb + (size_t)iei[bn * KNN + k] * C;
    bf16_t* er  = E  + t * (size_t)(4 * C);
    bf16_t* e2r = E2 + t * (size_t)(4 * C);
    for (int c = 0; c < C; ++c) {
        float cen = fc[c];
        float a = fe[c] - cen;
        float d = fi[c] - cen;
        er[c]         = (bf16_t)a;   e2r[c]         = (bf16_t)(a * a);
        er[C + c]     = (bf16_t)cen; e2r[C + c]     = (bf16_t)(cen * cen);
        er[2 * C + c] = (bf16_t)d;   e2r[2 * C + c] = (bf16_t)(d * d);
        er[3 * C + c] = (bf16_t)cen; e2r[3 * C + c] = (bf16_t)(cen * cen);
    }
}

// BN(batch stats) + LeakyReLU + max over k -> X[B,Np,Cout]
__global__ void qgs_qgscm_finalize(const float* __restrict__ T, const float* __restrict__ stats,
                                   const float* __restrict__ gamma, const float* __restrict__ beta,
                                   float* __restrict__ X, int Np, int Cout) {
    size_t t = (size_t)blockIdx.x * blockDim.x + threadIdx.x;
    if (t >= (size_t)BB * Np * Cout) return;
    int c = t % Cout;
    size_t bn = t / Cout;
    float Mtot = (float)BB * Np * KNN;
    float mean = stats[c] / Mtot;
    float var  = stats[1024 + c] / Mtot - mean * mean;
    float sc   = gamma[c] * rsqrtf(var + EPS_BN);
    float bt   = beta[c];
    const float* tr = T + bn * (size_t)KNN * Cout + c;
    float best = -3.0e38f;
    for (int k = 0; k < KNN; ++k) {
        float y = (tr[(size_t)k * Cout] - mean) * sc + bt;
        y = y > 0.f ? y : NEG * y;
        best = fmaxf(best, y);
    }
    X[t] = best;
}

// elementwise BN + LeakyReLU in place (stats over Mrows)
__global__ void qgs_bn_lrelu(float* __restrict__ y, const float* __restrict__ stats,
                             const float* __restrict__ gamma, const float* __restrict__ beta,
                             int Mrows, int C) {
    size_t t = (size_t)blockIdx.x * blockDim.x + threadIdx.x;
    if (t >= (size_t)Mrows * C) return;
    int c = t % C;
    float mean = stats[c] / (float)Mrows;
    float var  = stats[1024 + c] / (float)Mrows - mean * mean;
    float v = (y[t] - mean) * gamma[c] * rsqrtf(var + EPS_BN) + beta[c];
    y[t] = v > 0.f ? v : NEG * v;
}

// furthest point sampling; one block per batch, dmin in LDS
__global__ __launch_bounds__(256) void qgs_fps(const float* __restrict__ pts,
                                               int* __restrict__ out, int Np, int ns) {
    __shared__ float dmin[N1];
    __shared__ float rb[256];
    __shared__ int   ri[256];
    __shared__ int   s_last;
    int b = blockIdx.x, tid = threadIdx.x;
    const float* pb = pts + (size_t)b * Np * 3;
    for (int i = tid; i < Np; i += 256) dmin[i] = 1e10f;
    if (tid == 0) { s_last = 0; out[(size_t)b * ns] = 0; }
    __syncthreads();
    for (int s = 1; s < ns; ++s) {
        int last = s_last;
        float lx = pb[last * 3 + 0], ly = pb[last * 3 + 1], lz = pb[last * 3 + 2];
        float bestd = -1.f; int besti = 0;
        for (int i = tid; i < Np; i += 256) {
            float dx = pb[i * 3 + 0] - lx;
            float dy = pb[i * 3 + 1] - ly;
            float dz = pb[i * 3 + 2] - lz;
            float d = dx * dx + dy * dy + dz * dz;
            float dm = fminf(dmin[i], d);
            dmin[i] = dm;
            if (dm > bestd) { bestd = dm; besti = i; }
        }
        rb[tid] = bestd; ri[tid] = besti;
        __syncthreads();
        for (int off = 128; off > 0; off >>= 1) {
            if (tid < off && rb[tid + off] > rb[tid]) { rb[tid] = rb[tid + off]; ri[tid] = ri[tid + off]; }
            __syncthreads();
        }
        if (tid == 0) { s_last = ri[0]; out[(size_t)b * ns + s] = ri[0]; }
        __syncthreads();
    }
}

__global__ void qgs_gather(const float* __restrict__ f, const int* __restrict__ idx,
                           float* __restrict__ out, int NinPts, int NoutPts, int C) {
    size_t t = (size_t)blockIdx.x * blockDim.x + threadIdx.x;
    if (t >= (size_t)BB * NoutPts * C) return;
    int c = t % C;
    size_t bi = t / C;
    int b = bi / NoutPts, i = bi % NoutPts;
    out[t] = f[((size_t)b * NinPts + idx[(size_t)b * NoutPts + i]) * C + c];
}

// bottleneck concat [x1d(16) | x2d(64) | x3(256)] -> bf16 [BB*N3, 352]
__global__ void qgs_build_fcat(const float* __restrict__ x1d, const float* __restrict__ x2d,
                               const float* __restrict__ x3, bf16_t* __restrict__ out) {
    size_t t = (size_t)blockIdx.x * blockDim.x + threadIdx.x;
    if (t >= (size_t)BB * N3 * 352) return;
    int c = t % 352;
    size_t bn = t / 352;
    float v = 0.f;
    if (c < 16)       v = x1d[bn * 16 + c];
    else if (c < 80)  v = x2d[bn * 64 + (c - 16)];
    else if (c < 336) v = x3[bn * 256 + (c - 80)];
    out[t] = (bf16_t)v;
}

// generic cast+pad f32[Rin,Cin] -> bf16[Rpad,Cpad] (zeros outside)
__global__ void qgs_cast_pad(const float* __restrict__ in, bf16_t* __restrict__ out,
                             int Rin, int Cin, int Rpad, int Cpad) {
    size_t t = (size_t)blockIdx.x * blockDim.x + threadIdx.x;
    if (t >= (size_t)Rpad * Cpad) return;
    int c = t % Cpad;
    size_t r = t / Cpad;
    out[t] = (r < (size_t)Rin && c < Cin) ? (bf16_t)in[r * Cin + c] : (bf16_t)0.f;
}

__global__ void qgs_zero(float* __restrict__ p, int n) {
    int t = blockIdx.x * blockDim.x + threadIdx.x;
    if (t < n) p[t] = 0.f;
}

// three-NN + inverse-distance weights
__global__ void qgs_three_nn(const float* __restrict__ unk, const float* __restrict__ kno,
                             int* __restrict__ idx3, float* __restrict__ w3, int Mu, int S) {
    int t = blockIdx.x * blockDim.x + threadIdx.x;
    if (t >= BB * Mu) return;
    int b = t / Mu, m = t - b * Mu;
    const float* ub = unk + (size_t)b * Mu * 3;
    const float* kb = kno + (size_t)b * S * 3;
    float px = ub[m * 3 + 0], py = ub[m * 3 + 1], pz = ub[m * 3 + 2];
    float d0 = 3e38f, d1 = 3e38f, d2 = 3e38f;
    int i0 = 0, i1 = 0, i2 = 0;
    for (int s = 0; s < S; ++s) {
        float dx = kb[s * 3 + 0] - px;
        float dy = kb[s * 3 + 1] - py;
        float dz = kb[s * 3 + 2] - pz;
        float d = dx * dx + dy * dy + dz * dz;
        if (d < d0)      { d2 = d1; i2 = i1; d1 = d0; i1 = i0; d0 = d; i0 = s; }
        else if (d < d1) { d2 = d1; i2 = i1; d1 = d; i1 = s; }
        else if (d < d2) { d2 = d; i2 = s; }
    }
    float w0 = 1.f / (sqrtf(fmaxf(d0, 0.f)) + 1e-8f);
    float w1 = 1.f / (sqrtf(fmaxf(d1, 0.f)) + 1e-8f);
    float w2 = 1.f / (sqrtf(fmaxf(d2, 0.f)) + 1e-8f);
    float ws = 1.f / (w0 + w1 + w2);
    idx3[(size_t)t * 3 + 0] = i0; idx3[(size_t)t * 3 + 1] = i1; idx3[(size_t)t * 3 + 2] = i2;
    w3[(size_t)t * 3 + 0] = w0 * ws; w3[(size_t)t * 3 + 1] = w1 * ws; w3[(size_t)t * 3 + 2] = w2 * ws;
}

__global__ void qgs_three_interp(const float* __restrict__ feats, const int* __restrict__ idx3,
                                 const float* __restrict__ w3, float* __restrict__ out,
                                 int Mu, int S, int C) {
    size_t t = (size_t)blockIdx.x * blockDim.x + threadIdx.x;
    if (t >= (size_t)BB * Mu * C) return;
    int c = t % C;
    size_t bm = t / C;
    int b = bm / Mu;
    const float* fb = feats + (size_t)b * S * C;
    const int* id = idx3 + bm * 3;
    const float* w = w3 + bm * 3;
    out[t] = w[0] * fb[(size_t)id[0] * C + c]
           + w[1] * fb[(size_t)id[1] * C + c]
           + w[2] * fb[(size_t)id[2] * C + c];
}

// y7 [BB*N1,16] -> out [B,13,N]
__global__ void qgs_final_out(const float* __restrict__ y7, float* __restrict__ out) {
    size_t t = (size_t)blockIdx.x * blockDim.x + threadIdx.x;
    if (t >= (size_t)BB * 13 * N1) return;
    int n = t % N1;
    size_t bc = t / N1;
    int b = bc / 13, c = bc % 13;
    out[t] = y7[((size_t)b * N1 + n) * 16 + c];
}

// ======================= host orchestration ================================
static inline dim3 g1(size_t total, int bs = 256) { return dim3((unsigned)((total + bs - 1) / bs)); }

extern "C" void kernel_launch(void* const* d_in, const int* in_sizes, int n_in,
                              void* d_out, int out_size, void* d_ws, size_t ws_size,
                              hipStream_t stream) {
    const float* x    = (const float*)d_in[0];
    const float* w1r  = (const float*)d_in[1];
    const float* w1g  = (const float*)d_in[2];
    const float* b1g  = (const float*)d_in[3];
    const float* w1b  = (const float*)d_in[4];
    const float* bn1g = (const float*)d_in[5];
    const float* bn1b = (const float*)d_in[6];
    const float* w2r  = (const float*)d_in[7];
    const float* w2g  = (const float*)d_in[8];
    const float* b2g  = (const float*)d_in[9];
    const float* w2b  = (const float*)d_in[10];
    const float* bn2g = (const float*)d_in[11];
    const float* bn2b = (const float*)d_in[12];
    const float* w3r  = (const float*)d_in[13];
    const float* w3g  = (const float*)d_in[14];
    const float* b3g  = (const float*)d_in[15];
    const float* w3b  = (const float*)d_in[16];
    const float* bn3g = (const float*)d_in[17];
    const float* bn3b = (const float*)d_in[18];
    const float* w4   = (const float*)d_in[19];
    const float* bn4g = (const float*)d_in[20];
    const float* bn4b = (const float*)d_in[21];
    const float* w5   = (const float*)d_in[22];
    const float* bn5g = (const float*)d_in[23];
    const float* bn5b = (const float*)d_in[24];
    const float* w6   = (const float*)d_in[25];
    const float* bn6g = (const float*)d_in[26];
    const float* bn6b = (const float*)d_in[27];
    const float* w7   = (const float*)d_in[28];
    (void)in_sizes; (void)n_in; (void)out_size; (void)ws_size;

    constexpr int M1 = BB * N1 * KNN;   // 327680
    constexpr int M2 = BB * N2 * KNN;   // 163840
    constexpr int M3 = BB * N3 * KNN;   // 81920

    char* ws = (char*)d_ws;
    size_t off = 0;
    auto alloc = [&](size_t bytes) -> char* {
        char* p = ws + off;
        off += (bytes + 255) & ~(size_t)255;
        return p;
    };

    float*  pts1   = (float*)alloc((size_t)BB * N1 * 3 * 4);
    float*  eig1   = (float*)alloc((size_t)BB * N1 * 3 * 4);
    int*    ieu1   = (int*)  alloc((size_t)BB * N1 * KNN * 4);
    int*    iei1   = (int*)  alloc((size_t)BB * N1 * KNN * 4);
    bf16_t* e1     = (bf16_t*)alloc((size_t)M1 * 32 * 2);
    bf16_t* e1s    = (bf16_t*)alloc((size_t)M1 * 32 * 2);
    float*  t1     = (float*)alloc((size_t)M1 * 16 * 4);
    float*  x1     = (float*)alloc((size_t)BB * N1 * 16 * 4);
    float*  stats  = (float*)alloc(2048 * 4);
    int*    fps2i  = (int*)  alloc((size_t)BB * N2 * 4);
    float*  pts2   = (float*)alloc((size_t)BB * N2 * 3 * 4);
    float*  x1d2   = (float*)alloc((size_t)BB * N2 * 16 * 4);
    float*  eig2   = (float*)alloc((size_t)BB * N2 * 3 * 4);
    int*    ieu2   = (int*)  alloc((size_t)BB * N2 * KNN * 4);
    int*    iei2   = (int*)  alloc((size_t)BB * N2 * KNN * 4);
    bf16_t* e2     = (bf16_t*)alloc((size_t)M2 * 64 * 2);
    bf16_t* e2s    = (bf16_t*)alloc((size_t)M2 * 64 * 2);
    float*  t2     = (float*)alloc((size_t)M2 * 64 * 4);
    float*  x2     = (float*)alloc((size_t)BB * N2 * 64 * 4);
    int*    fps3i  = (int*)  alloc((size_t)BB * N3 * 4);
    float*  pts3   = (float*)alloc((size_t)BB * N3 * 3 * 4);
    float*  x2d3   = (float*)alloc((size_t)BB * N3 * 64 * 4);
    float*  x1d3   = (float*)alloc((size_t)BB * N3 * 16 * 4);
    float*  eig3   = (float*)alloc((size_t)BB * N3 * 3 * 4);
    int*    ieu3   = (int*)  alloc((size_t)BB * N3 * KNN * 4);
    int*    iei3   = (int*)  alloc((size_t)BB * N3 * KNN * 4);
    bf16_t* e3     = (bf16_t*)alloc((size_t)M3 * 256 * 2);
    bf16_t* e3s    = (bf16_t*)alloc((size_t)M3 * 256 * 2);
    float*  t3     = (float*)alloc((size_t)M3 * 256 * 4);
    float*  x3     = (float*)alloc((size_t)BB * N3 * 256 * 4);
    bf16_t* fcath  = (bf16_t*)alloc((size_t)BB * N3 * 352 * 2);
    float*  y4     = (float*)alloc((size_t)BB * N3 * 1024 * 4);
    int*    idx3a  = (int*)  alloc((size_t)BB * N2 * 3 * 4);
    float*  wi3a   = (float*)alloc((size_t)BB * N2 * 3 * 4);
    float*  fmid   = (float*)alloc((size_t)BB * N2 * 1024 * 4);
    int*    idx3b  = (int*)  alloc((size_t)BB * N1 * 3 * 4);
    float*  wi3b   = (float*)alloc((size_t)BB * N1 * 3 * 4);
    float*  fup    = (float*)alloc((size_t)BB * N1 * 1024 * 4);
    bf16_t* fuph   = (bf16_t*)alloc((size_t)BB * N1 * 1024 * 2);
    float*  y5     = (float*)alloc((size_t)BB * N1 * 512 * 4);
    bf16_t* y5h    = (bf16_t*)alloc((size_t)BB * N1 * 512 * 2);
    float*  y6     = (float*)alloc((size_t)BB * N1 * 256 * 4);
    bf16_t* y6h    = (bf16_t*)alloc((size_t)BB * N1 * 256 * 2);
    float*  y7     = (float*)alloc((size_t)BB * N1 * 16 * 4);
    bf16_t* w1rh   = (bf16_t*)alloc(16 * 32 * 2);
    bf16_t* w1gh   = (bf16_t*)alloc(16 * 32 * 2);
    bf16_t* w1bh   = (bf16_t*)alloc(16 * 32 * 2);
    bf16_t* w2rh   = (bf16_t*)alloc(64 * 64 * 2);
    bf16_t* w2gh   = (bf16_t*)alloc(64 * 64 * 2);
    bf16_t* w2bh   = (bf16_t*)alloc(64 * 64 * 2);
    bf16_t* w3rh   = (bf16_t*)alloc(256 * 256 * 2);
    bf16_t* w3gh   = (bf16_t*)alloc(256 * 256 * 2);
    bf16_t* w3bh   = (bf16_t*)alloc(256 * 256 * 2);
    bf16_t* w4h    = (bf16_t*)alloc((size_t)1024 * 352 * 2);
    bf16_t* w5h    = (bf16_t*)alloc((size_t)512 * 1024 * 2);
    bf16_t* w6h    = (bf16_t*)alloc((size_t)256 * 512 * 2);
    bf16_t* w7h    = (bf16_t*)alloc(16 * 256 * 2);

    // -------- weight conversion (bf16, padded) --------
    qgs_cast_pad<<<g1(16 * 32),          256, 0, stream>>>(w1r, w1rh, 16, 13, 16, 32);
    qgs_cast_pad<<<g1(16 * 32),          256, 0, stream>>>(w1g, w1gh, 16, 13, 16, 32);
    qgs_cast_pad<<<g1(16 * 32),          256, 0, stream>>>(w1b, w1bh, 16, 13, 16, 32);
    qgs_cast_pad<<<g1(64 * 64),          256, 0, stream>>>(w2r, w2rh, 64, 64, 64, 64);
    qgs_cast_pad<<<g1(64 * 64),          256, 0, stream>>>(w2g, w2gh, 64, 64, 64, 64);
    qgs_cast_pad<<<g1(64 * 64),          256, 0, stream>>>(w2b, w2bh, 64, 64, 64, 64);
    qgs_cast_pad<<<g1(256 * 256),        256, 0, stream>>>(w3r, w3rh, 256, 256, 256, 256);
    qgs_cast_pad<<<g1(256 * 256),        256, 0, stream>>>(w3g, w3gh, 256, 256, 256, 256);
    qgs_cast_pad<<<g1(256 * 256),        256, 0, stream>>>(w3b, w3bh, 256, 256, 256, 256);
    qgs_cast_pad<<<g1((size_t)1024*352), 256, 0, stream>>>(w4, w4h, 1024, 336, 1024, 352);
    qgs_cast_pad<<<g1((size_t)512*1024), 256, 0, stream>>>(w5, w5h, 512, 1024, 512, 1024);
    qgs_cast_pad<<<g1((size_t)256*512),  256, 0, stream>>>(w6, w6h, 256, 512, 256, 512);
    qgs_cast_pad<<<g1(16 * 256),         256, 0, stream>>>(w7, w7h, 13, 256, 16, 256);

    // -------- layer 1 --------
    qgs_extract_pts<<<g1((size_t)BB * N1 * 3), 256, 0, stream>>>(x, pts1);
    qgs_knn<<<g1(BB * N1), 256, 0, stream>>>(pts1, ieu1, N1);
    qgs_cov_eig<<<g1(BB * N1), 256, 0, stream>>>(pts1, ieu1, eig1, N1);
    qgs_knn<<<g1(BB * N1), 256, 0, stream>>>(eig1, iei1, N1);
    qgs_build_edge1<<<g1((size_t)M1), 256, 0, stream>>>(pts1, eig1, ieu1, iei1, e1, e1s, N1);
    qgs_zero<<<g1(2048), 256, 0, stream>>>(stats, 2048);
    qgs_wmma_qgscm<<<dim3(M1 / 16, 1), 32, 0, stream>>>(e1, e1s, w1rh, w1gh, w1bh, b1g,
                                                        t1, stats, M1, 16, 32);
    qgs_qgscm_finalize<<<g1((size_t)BB * N1 * 16), 256, 0, stream>>>(t1, stats, bn1g, bn1b,
                                                                     x1, N1, 16);
    // -------- FPS -> N/2 --------
    qgs_fps<<<BB, 256, 0, stream>>>(pts1, fps2i, N1, N2);
    qgs_gather<<<g1((size_t)BB * N2 * 3),  256, 0, stream>>>(pts1, fps2i, pts2, N1, N2, 3);
    qgs_gather<<<g1((size_t)BB * N2 * 16), 256, 0, stream>>>(x1, fps2i, x1d2, N1, N2, 16);

    // -------- layer 2 --------
    qgs_knn<<<g1(BB * N2), 256, 0, stream>>>(pts2, ieu2, N2);
    qgs_cov_eig<<<g1(BB * N2), 256, 0, stream>>>(pts2, ieu2, eig2, N2);
    qgs_knn<<<g1(BB * N2), 256, 0, stream>>>(eig2, iei2, N2);
    qgs_build_edge_group<<<g1((size_t)M2), 256, 0, stream>>>(x1d2, ieu2, iei2, e2, e2s, N2, 16);
    qgs_zero<<<g1(2048), 256, 0, stream>>>(stats, 2048);
    qgs_wmma_qgscm<<<dim3(M2 / 16, 4), 32, 0, stream>>>(e2, e2s, w2rh, w2gh, w2bh, b2g,
                                                        t2, stats, M2, 64, 64);
    qgs_qgscm_finalize<<<g1((size_t)BB * N2 * 64), 256, 0, stream>>>(t2, stats, bn2g, bn2b,
                                                                     x2, N2, 64);
    // -------- FPS -> N/4 --------
    qgs_fps<<<BB, 256, 0, stream>>>(pts2, fps3i, N2, N3);
    qgs_gather<<<g1((size_t)BB * N3 * 3),  256, 0, stream>>>(pts2, fps3i, pts3, N2, N3, 3);
    qgs_gather<<<g1((size_t)BB * N3 * 64), 256, 0, stream>>>(x2, fps3i, x2d3, N2, N3, 64);
    qgs_gather<<<g1((size_t)BB * N3 * 16), 256, 0, stream>>>(x1d2, fps3i, x1d3, N2, N3, 16);

    // -------- layer 3 --------
    qgs_knn<<<g1(BB * N3), 256, 0, stream>>>(pts3, ieu3, N3);
    qgs_cov_eig<<<g1(BB * N3), 256, 0, stream>>>(pts3, ieu3, eig3, N3);
    qgs_knn<<<g1(BB * N3), 256, 0, stream>>>(eig3, iei3, N3);
    qgs_build_edge_group<<<g1((size_t)M3), 256, 0, stream>>>(x2d3, ieu3, iei3, e3, e3s, N3, 64);
    qgs_zero<<<g1(2048), 256, 0, stream>>>(stats, 2048);
    qgs_wmma_qgscm<<<dim3(M3 / 16, 16), 32, 0, stream>>>(e3, e3s, w3rh, w3gh, w3bh, b3g,
                                                         t3, stats, M3, 256, 256);
    qgs_qgscm_finalize<<<g1((size_t)BB * N3 * 256), 256, 0, stream>>>(t3, stats, bn3g, bn3b,
                                                                      x3, N3, 256);

    // -------- bottleneck (w4) --------
    qgs_build_fcat<<<g1((size_t)BB * N3 * 352), 256, 0, stream>>>(x1d3, x2d3, x3, fcath);
    qgs_zero<<<g1(2048), 256, 0, stream>>>(stats, 2048);
    qgs_wmma_gemm<true><<<dim3(BB * N3 / 16, 1024 / 16), 32, 0, stream>>>(
        fcath, w4h, y4, stats, BB * N3, 1024, 352);
    qgs_bn_lrelu<<<g1((size_t)BB * N3 * 1024), 256, 0, stream>>>(y4, stats, bn4g, bn4b,
                                                                 BB * N3, 1024);

    // -------- interpolate up: N/4 -> N/2 -> N --------
    qgs_three_nn<<<g1(BB * N2), 256, 0, stream>>>(pts2, pts3, idx3a, wi3a, N2, N3);
    qgs_three_interp<<<g1((size_t)BB * N2 * 1024), 256, 0, stream>>>(y4, idx3a, wi3a,
                                                                     fmid, N2, N3, 1024);
    qgs_three_nn<<<g1(BB * N1), 256, 0, stream>>>(pts1, pts2, idx3b, wi3b, N1, N2);
    qgs_three_interp<<<g1((size_t)BB * N1 * 1024), 256, 0, stream>>>(fmid, idx3b, wi3b,
                                                                     fup, N1, N2, 1024);

    // -------- head MLPs (w5, w6, w7) --------
    qgs_cast_pad<<<g1((size_t)BB * N1 * 1024), 256, 0, stream>>>(fup, fuph,
                                                                 BB * N1, 1024, BB * N1, 1024);
    qgs_zero<<<g1(2048), 256, 0, stream>>>(stats, 2048);
    qgs_wmma_gemm<true><<<dim3(BB * N1 / 16, 512 / 16), 32, 0, stream>>>(
        fuph, w5h, y5, stats, BB * N1, 512, 1024);
    qgs_bn_lrelu<<<g1((size_t)BB * N1 * 512), 256, 0, stream>>>(y5, stats, bn5g, bn5b,
                                                                BB * N1, 512);
    qgs_cast_pad<<<g1((size_t)BB * N1 * 512), 256, 0, stream>>>(y5, y5h,
                                                                BB * N1, 512, BB * N1, 512);
    qgs_zero<<<g1(2048), 256, 0, stream>>>(stats, 2048);
    qgs_wmma_gemm<true><<<dim3(BB * N1 / 16, 256 / 16), 32, 0, stream>>>(
        y5h, w6h, y6, stats, BB * N1, 256, 512);
    qgs_bn_lrelu<<<g1((size_t)BB * N1 * 256), 256, 0, stream>>>(y6, stats, bn6g, bn6b,
                                                                BB * N1, 256);
    qgs_cast_pad<<<g1((size_t)BB * N1 * 256), 256, 0, stream>>>(y6, y6h,
                                                                BB * N1, 256, BB * N1, 256);
    qgs_wmma_gemm<false><<<dim3(BB * N1 / 16, 1), 32, 0, stream>>>(
        y6h, w7h, y7, nullptr, BB * N1, 16, 256);

    qgs_final_out<<<g1((size_t)BB * 13 * N1), 256, 0, stream>>>(y7, (float*)d_out);
}
